// cVAE_84413287235758
// MI455X (gfx1250) — matrix operand order repaired and checked by reading, not compiled, not run on hardware
//
#include <hip/hip_runtime.h>
#include <hip/hip_bf16.h>
#include <stdint.h>

#define B_    128
#define PAST_ 256
#define FUT_  128
#define D_    128
#define RU_   512
#define LD_   128
#define FCN_  512
#define TL_   128
#define HDEC_ 640

typedef __attribute__((ext_vector_type(16))) __bf16 v16bf;
typedef __attribute__((ext_vector_type(8)))  __bf16 v8bf;
typedef __attribute__((ext_vector_type(8)))  float  v8f;

// float -> bf16 round-to-nearest-even (pure bit ops)
__device__ __forceinline__ __bf16 f2bf(float x) {
  union { float f; unsigned u; } a; a.f = x;
  unsigned r = (a.u + 0x7FFFu + ((a.u >> 16) & 1u)) >> 16;
  union { unsigned short s; __bf16 b; } o; o.s = (unsigned short)r;
  return o.b;
}

// Async global->LDS copy, 16B per lane (ASYNCcnt-tracked; gfx1250 path)
__device__ __forceinline__ void async_copy_16B(const __bf16* gsrc, __bf16* lds_dst) {
  unsigned lds_addr = (unsigned)(uintptr_t)lds_dst;   // low 32 bits of flat LDS addr = LDS byte offset
  asm volatile("global_load_async_to_lds_b128 %0, %1, off"
               :: "v"(lds_addr), "v"(gsrc) : "memory");
}
__device__ __forceinline__ void wait_async() {
  asm volatile("s_wait_asynccnt 0x0" ::: "memory");
}

// A fragment (16x32 MxK) from an LDS tile with leading dim ldk (row0 == 0).
// Layout: lanes 0-15 hold K in {0..15}, lanes 16-31 hold {16..31}.
__device__ __forceinline__ v16bf lds_a_frag(const __bf16* sA, int ldk, int kb, int lane) {
  int l15 = lane & 15, hi = lane >> 4;
  const __bf16* p = sA + l15 * ldk + kb + hi * 16;
  union { v16bf v; v8bf h[2]; } u;
  u.h[0] = *(const v8bf*)p;
  u.h[1] = *(const v8bf*)(p + 8);
  return u.v;
}

// B fragment (32x16 KxN): B[k][n] = W[n][k], W row-major (N x ldk).
// Lane n holds contiguous 16 K-elements of W row n.
__device__ __forceinline__ v16bf load_b_frag(const __bf16* W, int ldk, int nrow0, int kb, int lane) {
  int l15 = lane & 15, hi = lane >> 4;
  return *(const v16bf*)(W + (size_t)(nrow0 + l15) * ldk + kb + hi * 16);
}

__device__ __forceinline__ v8f wmma_bf16(v16bf a, v16bf b, v8f c) {
  return __builtin_amdgcn_wmma_f32_16x16x32_bf16(false, a, false, b, (short)0, c, false, false);
}

__device__ __forceinline__ float sigm(float x) { return 1.0f / (1.0f + expf(-x)); }

// ---------------------------------------------------------------------------
__global__ void k_cvt(const float* __restrict__ s, __bf16* __restrict__ d, int n) {
  for (int i = blockIdx.x * blockDim.x + threadIdx.x; i < n; i += gridDim.x * blockDim.x)
    d[i] = f2bf(s[i]);
}

// ---------------------------------------------------------------------------
// GEMM: out[128 x N] = act(A(128xK,bf16) @ W(NxK,bf16)^T + bias).
// 128 threads = 4 waves per block; all 4 waves share one M-tile staged in LDS.
// act: 0=none, 1=leaky_relu(0.01), 2=relu.
template<int K>
__global__ void k_gemm(const __bf16* __restrict__ A, int lda, int abase,
                       const __bf16* __restrict__ W, int N,
                       const float* __restrict__ bias, int act,
                       float* __restrict__ outf, int ldo_f, int obase_f,
                       __bf16* __restrict__ outb, int ldo_b) {
  __shared__ __bf16 sA[16 * K];
  int tid = threadIdx.x, lane = tid & 31, wid = tid >> 5;
  int ntiles = N >> 4;
  int wave = blockIdx.x * 4 + wid;
  int nt = wave % ntiles, mt = wave / ntiles;
  int m0 = ((blockIdx.x * 4) / ntiles) * 16;          // block-uniform (== mt*16)
  const __bf16* Ab = A + abase;

  // cooperative stage of the A M-tile (load + ds_store path)
#pragma unroll
  for (int c = tid; c < 16 * K / 8; c += 128) {
    int row = c / (K / 8), off = (c % (K / 8)) * 8;
    *(v8bf*)(sA + row * K + off) = *(const v8bf*)(Ab + (size_t)(m0 + row) * lda + off);
  }
  __syncthreads();

  v8f acc = {};
#pragma unroll
  for (int k = 0; k < K; k += 32)
    acc = wmma_bf16(lds_a_frag(sA, K, k, lane), load_b_frag(W, K, nt * 16, k, lane), acc);

  int l15 = lane & 15, hi = lane >> 4;
  int n = nt * 16 + l15;
  float bv = bias ? bias[n] : 0.0f;
#pragma unroll
  for (int v = 0; v < 8; v++) {
    int m = mt * 16 + v + 8 * hi;
    float x = acc[v] + bv;
    if (act == 1) x = (x > 0.0f) ? x : 0.01f * x;
    else if (act == 2) x = (x > 0.0f) ? x : 0.0f;
    if (outf) outf[(size_t)m * ldo_f + obase_f + n] = x;
    if (outb) outb[(size_t)m * ldo_b + n] = f2bf(x);
  }
}

// ---------------------------------------------------------------------------
// One GRU step. 4 waves/block share one M-tile; X and h tiles staged in LDS
// (h via async global->LDS DMA). n-gate needs gi/gh split: 6 accumulators.
template<int RU>
__global__ void k_gru_step(const __bf16* __restrict__ X, int ldx, int xbase,
                           const __bf16* __restrict__ Wih,   // (3RU x 128) bf16
                           const __bf16* __restrict__ Whh,   // (3RU x RU) bf16
                           const float* __restrict__ bih, const float* __restrict__ bhh,
                           const float* __restrict__ h_in_f, const __bf16* __restrict__ h_in_b,
                           float* __restrict__ h_out_f, __bf16* __restrict__ h_out_b) {
  constexpr int DX = 128;
  constexpr int NT = RU / 16;
  __shared__ __bf16 sX[16 * DX];
  __shared__ __bf16 sH[16 * RU];
  int tid = threadIdx.x, lane = tid & 31, wid = tid >> 5;
  int wave = blockIdx.x * 4 + wid;
  int nh = wave % NT, mt = wave / NT;
  int m0 = ((blockIdx.x * 4) / NT) * 16;              // block-uniform (== mt*16)

  // stage X tile: plain load + ds_store
#pragma unroll
  for (int c = tid; c < 16 * DX / 8; c += 128) {
    int row = c / (DX / 8), off = (c % (DX / 8)) * 8;
    *(v8bf*)(sX + row * DX + off) =
        *(const v8bf*)(X + xbase + (size_t)(m0 + row) * ldx + off);
  }
  // stage h tile: async DMA global->LDS
#pragma unroll
  for (int c = tid; c < 16 * RU / 8; c += 128) {
    int row = c / (RU / 8), off = (c % (RU / 8)) * 8;
    async_copy_16B(h_in_b + (size_t)(m0 + row) * RU + off, sH + row * RU + off);
  }
  wait_async();
  __syncthreads();

  v8f a_ir = {}, a_iz = {}, a_in = {}, a_hr = {}, a_hz = {}, a_hn = {};
#pragma unroll
  for (int k = 0; k < DX; k += 32) {
    v16bf a = lds_a_frag(sX, DX, k, lane);
    a_ir = wmma_bf16(a, load_b_frag(Wih, DX, 0 * RU + nh * 16, k, lane), a_ir);
    a_iz = wmma_bf16(a, load_b_frag(Wih, DX, 1 * RU + nh * 16, k, lane), a_iz);
    a_in = wmma_bf16(a, load_b_frag(Wih, DX, 2 * RU + nh * 16, k, lane), a_in);
  }
#pragma unroll
  for (int k = 0; k < RU; k += 32) {
    v16bf a = lds_a_frag(sH, RU, k, lane);
    a_hr = wmma_bf16(a, load_b_frag(Whh, RU, 0 * RU + nh * 16, k, lane), a_hr);
    a_hz = wmma_bf16(a, load_b_frag(Whh, RU, 1 * RU + nh * 16, k, lane), a_hz);
    a_hn = wmma_bf16(a, load_b_frag(Whh, RU, 2 * RU + nh * 16, k, lane), a_hn);
  }

  int l15 = lane & 15, hi = lane >> 4;
  int nc = nh * 16 + l15;
  float b_ir = bih[nc], b_iz = bih[RU + nc], b_in = bih[2 * RU + nc];
  float c_hr = bhh[nc], c_hz = bhh[RU + nc], c_hn = bhh[2 * RU + nc];
#pragma unroll
  for (int v = 0; v < 8; v++) {
    int m = mt * 16 + v + 8 * hi;
    float r  = sigm(a_ir[v] + b_ir + a_hr[v] + c_hr);
    float z  = sigm(a_iz[v] + b_iz + a_hz[v] + c_hz);
    float nn = tanhf(a_in[v] + b_in + r * (a_hn[v] + c_hn));
    float ho = h_in_f[(size_t)m * RU + nc];
    float hv = (1.0f - z) * nn + z * ho;
    h_out_f[(size_t)m * RU + nc] = hv;
    h_out_b[(size_t)m * RU + nc] = f2bf(hv);
  }
}

// ---------------------------------------------------------------------------
__global__ void k_concat(const __bf16* __restrict__ hp, const __bf16* __restrict__ hf,
                         __bf16* __restrict__ feat) {
  int i = blockIdx.x * blockDim.x + threadIdx.x;       // B*1024
  int b = i >> 10, c = i & 1023;
  feat[i] = (c < RU_) ? hp[b * RU_ + c] : hf[b * RU_ + (c - RU_)];
}

__global__ void k_build_dec(const float* __restrict__ zmu, const float* __restrict__ zlv,
                            const float* __restrict__ eps, const float* __restrict__ hp_f,
                            float* __restrict__ dh_f, __bf16* __restrict__ dh_b,
                            const float* __restrict__ past, __bf16* __restrict__ x0) {
  int i = blockIdx.x * blockDim.x + threadIdx.x;       // B * 768
  int b = i / 768, c = i % 768;
  if (c < HDEC_) {
    float v;
    if (c < RU_) v = hp_f[b * RU_ + c];
    else {
      int j = c - RU_;
      v = zmu[b * LD_ + j] + eps[b * LD_ + j] * expf(0.5f * zlv[b * LD_ + j]);
    }
    dh_f[b * HDEC_ + c] = v;
    dh_b[b * HDEC_ + c] = f2bf(v);
  } else {
    int d = c - HDEC_;
    x0[b * D_ + d] = f2bf(past[(size_t)b * PAST_ * D_ + (PAST_ - 1) * D_ + d]);
  }
}

// ---------------------------------------------------------------------------
template<int K>
static void launch_gemm(hipStream_t stream, const __bf16* A, int lda, int abase,
                        const __bf16* W, int N, const float* bias, int act,
                        float* outf, int ldo_f, int obase_f, __bf16* outb, int ldo_b) {
  int blocks = (8 * (N / 16)) / 4;
  k_gemm<K><<<blocks, 128, 0, stream>>>(A, lda, abase, W, N, bias, act,
                                        outf, ldo_f, obase_f, outb, ldo_b);
}

extern "C" void kernel_launch(void* const* d_in, const int* in_sizes, int n_in,
                              void* d_out, int out_size, void* d_ws, size_t ws_size,
                              hipStream_t stream) {
  (void)in_sizes; (void)n_in; (void)out_size; (void)ws_size;

  const float* past    = (const float*)d_in[0];
  const float* future  = (const float*)d_in[1];
  const float* eps     = (const float*)d_in[2];
  const float* phi_Wih = (const float*)d_in[3];
  const float* phi_Whh = (const float*)d_in[4];
  const float* phi_bih = (const float*)d_in[5];
  const float* phi_bhh = (const float*)d_in[6];
  const float* xr_Wih  = (const float*)d_in[7];
  const float* xr_Whh  = (const float*)d_in[8];
  const float* xr_bih  = (const float*)d_in[9];
  const float* xr_bhh  = (const float*)d_in[10];
  const float* mu_W1   = (const float*)d_in[11];
  const float* mu_b1   = (const float*)d_in[12];
  const float* mu_W2   = (const float*)d_in[13];
  const float* mu_b2   = (const float*)d_in[14];
  const float* mu_W3   = (const float*)d_in[15];
  const float* mu_b3   = (const float*)d_in[16];
  const float* lv_W1   = (const float*)d_in[17];
  const float* lv_b1   = (const float*)d_in[18];
  const float* lv_W2   = (const float*)d_in[19];
  const float* lv_b2   = (const float*)d_in[20];
  const float* lv_W3   = (const float*)d_in[21];
  const float* lv_b3   = (const float*)d_in[22];
  const float* dec_Wih = (const float*)d_in[23];
  const float* dec_Whh = (const float*)d_in[24];
  const float* dec_bih = (const float*)d_in[25];
  const float* dec_bhh = (const float*)d_in[26];
  const float* fc_W    = (const float*)d_in[27];
  const float* fc_b    = (const float*)d_in[28];
  const float* out_W   = (const float*)d_in[29];
  const float* out_bias= (const float*)d_in[30];

  float* outp    = (float*)d_out;
  float* zmu_out = outp + (size_t)B_ * TL_ * D_;
  float* zlv_out = zmu_out + (size_t)B_ * LD_;

  char* wsb = (char*)d_ws; size_t off = 0;
  auto alloc = [&](size_t bytes) -> void* {
    void* p = wsb + off; off += (bytes + 255) & ~(size_t)255; return p;
  };

  __bf16* past_b   = (__bf16*)alloc((size_t)B_ * PAST_ * D_ * 2);
  __bf16* fut_b    = (__bf16*)alloc((size_t)B_ * FUT_ * D_ * 2);
  __bf16* phiWih_b = (__bf16*)alloc((size_t)3 * RU_ * D_ * 2);
  __bf16* phiWhh_b = (__bf16*)alloc((size_t)3 * RU_ * RU_ * 2);
  __bf16* xrWih_b  = (__bf16*)alloc((size_t)3 * RU_ * D_ * 2);
  __bf16* xrWhh_b  = (__bf16*)alloc((size_t)3 * RU_ * RU_ * 2);
  __bf16* decWih_b = (__bf16*)alloc((size_t)3 * HDEC_ * D_ * 2);
  __bf16* decWhh_b = (__bf16*)alloc((size_t)3 * HDEC_ * HDEC_ * 2);
  __bf16* muW1_b   = (__bf16*)alloc((size_t)512 * 1024 * 2);
  __bf16* muW2_b   = (__bf16*)alloc((size_t)256 * 512 * 2);
  __bf16* muW3_b   = (__bf16*)alloc((size_t)LD_ * 256 * 2);
  __bf16* lvW1_b   = (__bf16*)alloc((size_t)512 * 1024 * 2);
  __bf16* lvW2_b   = (__bf16*)alloc((size_t)256 * 512 * 2);
  __bf16* lvW3_b   = (__bf16*)alloc((size_t)LD_ * 256 * 2);
  __bf16* fcW_b    = (__bf16*)alloc((size_t)FCN_ * HDEC_ * 2);
  __bf16* outW_b   = (__bf16*)alloc((size_t)D_ * FCN_ * 2);

  float*  hP0_f = (float*)alloc((size_t)B_ * RU_ * 4);
  float*  hP1_f = (float*)alloc((size_t)B_ * RU_ * 4);
  __bf16* hP0_b = (__bf16*)alloc((size_t)B_ * RU_ * 2);
  __bf16* hP1_b = (__bf16*)alloc((size_t)B_ * RU_ * 2);
  float*  hX0_f = (float*)alloc((size_t)B_ * RU_ * 4);
  float*  hX1_f = (float*)alloc((size_t)B_ * RU_ * 4);
  __bf16* hX0_b = (__bf16*)alloc((size_t)B_ * RU_ * 2);
  __bf16* hX1_b = (__bf16*)alloc((size_t)B_ * RU_ * 2);

  __bf16* feat_b = (__bf16*)alloc((size_t)B_ * 1024 * 2);
  __bf16* m1_b   = (__bf16*)alloc((size_t)B_ * 512 * 2);
  __bf16* m2_b   = (__bf16*)alloc((size_t)B_ * 256 * 2);

  float*  dh0_f = (float*)alloc((size_t)B_ * HDEC_ * 4);
  float*  dh1_f = (float*)alloc((size_t)B_ * HDEC_ * 4);
  __bf16* dh0_b = (__bf16*)alloc((size_t)B_ * HDEC_ * 2);
  __bf16* dh1_b = (__bf16*)alloc((size_t)B_ * HDEC_ * 2);
  __bf16* xcur_b = (__bf16*)alloc((size_t)B_ * D_ * 2);
  __bf16* fcout_b = (__bf16*)alloc((size_t)B_ * FCN_ * 2);

  auto cvt = [&](const float* s, __bf16* d, size_t n) {
    int blocks = (int)((n + 255) / 256); if (blocks > 2048) blocks = 2048;
    k_cvt<<<blocks, 256, 0, stream>>>(s, d, (int)n);
  };
  cvt(past, past_b, (size_t)B_ * PAST_ * D_);
  cvt(future, fut_b, (size_t)B_ * FUT_ * D_);
  cvt(phi_Wih, phiWih_b, (size_t)3 * RU_ * D_);
  cvt(phi_Whh, phiWhh_b, (size_t)3 * RU_ * RU_);
  cvt(xr_Wih, xrWih_b, (size_t)3 * RU_ * D_);
  cvt(xr_Whh, xrWhh_b, (size_t)3 * RU_ * RU_);
  cvt(dec_Wih, decWih_b, (size_t)3 * HDEC_ * D_);
  cvt(dec_Whh, decWhh_b, (size_t)3 * HDEC_ * HDEC_);
  cvt(mu_W1, muW1_b, (size_t)512 * 1024);
  cvt(mu_W2, muW2_b, (size_t)256 * 512);
  cvt(mu_W3, muW3_b, (size_t)LD_ * 256);
  cvt(lv_W1, lvW1_b, (size_t)512 * 1024);
  cvt(lv_W2, lvW2_b, (size_t)256 * 512);
  cvt(lv_W3, lvW3_b, (size_t)LD_ * 256);
  cvt(fc_W, fcW_b, (size_t)FCN_ * HDEC_);
  cvt(out_W, outW_b, (size_t)D_ * FCN_);

  hipMemsetAsync(hP0_f, 0, (size_t)B_ * RU_ * 4, stream);
  hipMemsetAsync(hP0_b, 0, (size_t)B_ * RU_ * 2, stream);
  hipMemsetAsync(hX0_f, 0, (size_t)B_ * RU_ * 4, stream);
  hipMemsetAsync(hX0_b, 0, (size_t)B_ * RU_ * 2, stream);

  // ---- phi GRU over past (256 steps) ----
  float* hin_f = hP0_f; float* hout_f = hP1_f;
  __bf16* hin_b = hP0_b; __bf16* hout_b = hP1_b;
  for (int t = 0; t < PAST_; t++) {
    k_gru_step<RU_><<<64, 128, 0, stream>>>(past_b, PAST_ * D_, t * D_,
        phiWih_b, phiWhh_b, phi_bih, phi_bhh,
        hin_f, hin_b, hout_f, hout_b);
    { float* tf = hin_f; hin_f = hout_f; hout_f = tf; }
    { __bf16* tb = hin_b; hin_b = hout_b; hout_b = tb; }
  }
  float* hpast_f = hin_f; __bf16* hpast_b = hin_b;

  // ---- xr GRU over future (128 steps) ----
  hin_f = hX0_f; hout_f = hX1_f; hin_b = hX0_b; hout_b = hX1_b;
  for (int t = 0; t < FUT_; t++) {
    k_gru_step<RU_><<<64, 128, 0, stream>>>(fut_b, FUT_ * D_, t * D_,
        xrWih_b, xrWhh_b, xr_bih, xr_bhh,
        hin_f, hin_b, hout_f, hout_b);
    { float* tf = hin_f; hin_f = hout_f; hout_f = tf; }
    { __bf16* tb = hin_b; hin_b = hout_b; hout_b = tb; }
  }
  __bf16* hfut_b = hin_b;

  // ---- features + MLPs ----
  k_concat<<<(B_ * 1024) / 256, 256, 0, stream>>>(hpast_b, hfut_b, feat_b);

  launch_gemm<1024>(stream, feat_b, 1024, 0, muW1_b, 512, mu_b1, 1, nullptr, 0, 0, m1_b, 512);
  launch_gemm<512>(stream, m1_b,    512, 0, muW2_b, 256, mu_b2, 1, nullptr, 0, 0, m2_b, 256);
  launch_gemm<256>(stream, m2_b,    256, 0, muW3_b, LD_, mu_b3, 0, zmu_out, LD_, 0, nullptr, 0);
  launch_gemm<1024>(stream, feat_b, 1024, 0, lvW1_b, 512, lv_b1, 1, nullptr, 0, 0, m1_b, 512);
  launch_gemm<512>(stream, m1_b,    512, 0, lvW2_b, 256, lv_b2, 1, nullptr, 0, 0, m2_b, 256);
  launch_gemm<256>(stream, m2_b,    256, 0, lvW3_b, LD_, lv_b3, 0, zlv_out, LD_, 0, nullptr, 0);

  // ---- reparameterize + decoder init ----
  k_build_dec<<<(B_ * 768) / 256, 256, 0, stream>>>(zmu_out, zlv_out, eps, hpast_f,
                                                    dh0_f, dh0_b, past, xcur_b);

  // ---- decoder loop (128 steps) ----
  float* din_f = dh0_f; float* dout_f = dh1_f;
  __bf16* din_b = dh0_b; __bf16* dout_b = dh1_b;
  for (int t = 0; t < TL_; t++) {
    k_gru_step<HDEC_><<<80, 128, 0, stream>>>(xcur_b, D_, 0,
        decWih_b, decWhh_b, dec_bih, dec_bhh,
        din_f, din_b, dout_f, dout_b);
    // fc: relu(h @ fc_W^T + fc_b) -> bf16
    launch_gemm<HDEC_>(stream, dout_b, HDEC_, 0, fcW_b, FCN_, fc_b, 2,
                       nullptr, 0, 0, fcout_b, FCN_);
    // out: -> x_mu[:, t, :] (f32) and next-step input (bf16)
    launch_gemm<FCN_>(stream, fcout_b, FCN_, 0, outW_b, D_, out_bias, 0,
                      outp, TL_ * D_, t * D_, xcur_b, D_);
    { float* tf = din_f; din_f = dout_f; dout_f = tf; }
    { __bf16* tb = din_b; din_b = dout_b; dout_b = tb; }
  }
}
